// TtMambaBlock_51153060496170
// MI455X (gfx1250) — matrix operand order repaired and checked by reading, not compiled, not run on hardware
//
#include <hip/hip_runtime.h>
#include <hip/hip_bf16.h>
#include <cstdint>
#include <cstddef>

typedef __bf16 bf16;
typedef __attribute__((ext_vector_type(16))) __bf16 v16bf;
typedef __attribute__((ext_vector_type(8)))  __bf16 v8bf;
typedef __attribute__((ext_vector_type(2)))  __bf16 v2bf;
typedef __attribute__((ext_vector_type(8)))  float  v8f;

#define H_DIM   2048
#define U_DIM   256
#define N_DIM   16
#define R_DIM   128
#define DIN_DIM 4096
#define DBC_W   (R_DIM + 2 * N_DIM)   // 160

enum Epi { EPI_NONE = 0, EPI_CONV_SILU = 1, EPI_SOFTPLUS = 2 };

// GEMM tiling: 128x64 per 256-thread workgroup (8 wave32, 4x2 wave grid),
// each wave holds a 32x32 patch = four 16x16 f32 accumulators.
// Double-buffered LDS: one barrier per K-step, global->reg prefetch overlaps WMMA.
constexpr int BM = 128, BN = 64, BK = 32;
constexpr int KPAD = 8;   // LDS row pad (bf16) -> 80B row stride, conflict-free b128 reads

template <int EPI, bool NGUARD>
__global__ __launch_bounds__(256) void gemm_bf16_wmma(
    const float* __restrict__ A, int lda,
    const float* __restrict__ B, int ldb,
    float* __restrict__ C, int ldc,
    int M, int N, int K,
    const float* __restrict__ ep0,   // conv_states (CONV) / dt_bias (SOFTPLUS)
    const float* __restrict__ ep1,   // conv_w  (4 x ldc)
    const float* __restrict__ ep2)   // conv_bias (ldc)
{
  __shared__ bf16 sA[2][BM][BK + KPAD];    // row-major A tile (bf16), ping-pong
  __shared__ bf16 sBt[2][BN][BK + KPAD];   // B tile transposed: sBt[buf][n][k]

  const int tid   = threadIdx.x;
  const int lane  = tid & 31;
  const int lrow  = lane & 15;   // M index (A frag) / N index (B frag) / column (C)
  const int khalf = lane >> 4;   // K-half selector per WMMA VGPR layout
  const int wave  = tid >> 5;
  const int waveM = (wave & 3) * 32;
  const int waveN = (wave >> 2) * 32;

  const int tileM = blockIdx.y * BM;
  const int tileN = blockIdx.x * BN;

  v8f acc[2][2] = {};

  // Cooperative staging maps:
  //  A tile 128x32 f32: 4 float4/thread (row a_r+32*rep, cols a_c..a_c+3)
  //  B tile  32x64 f32: each thread owns a 2(k) x 4(n) patch so the transposed
  //  commit packs (k,k+1) bf16 pairs -> single ds_store_b32 per n column.
  const int a_c  = (tid & 7) * 4;     // k column within A tile
  const int a_r  = tid >> 3;          // A row 0..31 (x4 reps)
  const int b_c  = (tid & 15) * 4;    // n column within B tile
  const int b_k  = (tid >> 4) * 2;    // k row pair 0,2,..,30

  float4 aReg[4];
  float4 bReg[2];                     // rows b_k and b_k+1, cols b_c..b_c+3

  // ---- load one K-tile (global f32) into registers ----
  auto load_tile = [&](int kt) {
#pragma unroll
    for (int rep = 0; rep < 4; ++rep) {
      const int r = a_r + rep * 32;
      aReg[rep] = *(const float4*)(A + (size_t)(tileM + r) * lda + (kt + a_c));
    }
#pragma unroll
    for (int rep = 0; rep < 2; ++rep) {
      const int kk = b_k + rep;
      if (NGUARD) {
        const int n0 = tileN + b_c;
        float4 v = make_float4(0.f, 0.f, 0.f, 0.f);
        if (n0 + 3 < N) {
          v = *(const float4*)(B + (size_t)(kt + kk) * ldb + n0);
        } else if (n0 < N) {
          const float* p = B + (size_t)(kt + kk) * ldb;
          v.x = p[n0];
          if (n0 + 1 < N) v.y = p[n0 + 1];
          if (n0 + 2 < N) v.z = p[n0 + 2];
        }
        bReg[rep] = v;
      } else {
        bReg[rep] = *(const float4*)(B + (size_t)(kt + kk) * ldb + (tileN + b_c));
      }
    }
  };

  // ---- convert registers to bf16 and commit into LDS buffer `buf` ----
  auto commit_tile = [&](int buf) {
#pragma unroll
    for (int rep = 0; rep < 4; ++rep) {
      const int r = a_r + rep * 32;
      bf16* d = &sA[buf][r][a_c];
      d[0] = (bf16)aReg[rep].x; d[1] = (bf16)aReg[rep].y;
      d[2] = (bf16)aReg[rep].z; d[3] = (bf16)aReg[rep].w;
    }
    // transpose commit: (k,k+1) pair at column n -> one packed b32 store
    const float c0[4] = {bReg[0].x, bReg[0].y, bReg[0].z, bReg[0].w};
    const float c1[4] = {bReg[1].x, bReg[1].y, bReg[1].z, bReg[1].w};
#pragma unroll
    for (int j = 0; j < 4; ++j) {
      v2bf p;
      p[0] = (bf16)c0[j];   // k   = b_k
      p[1] = (bf16)c1[j];   // k+1 = b_k+1
      *(v2bf*)&sBt[buf][b_c + j][b_k] = p;
    }
  };

  const int nIter = K / BK;
  load_tile(0);
  commit_tile(0);
  __syncthreads();

  for (int it = 0; it < nIter; ++it) {
    const int cur = it & 1;
    const bool hasNext = (it + 1 < nIter);

    // global->reg prefetch of next tile overlaps with WMMA on current tile
    if (hasNext) load_tile((it + 1) * BK);
    // L2 prefetch two tiles ahead (global_prefetch_b8)
    if (it + 2 < nIter) {
      __builtin_prefetch(A + (size_t)(tileM + a_r) * lda + ((it + 2) * BK + a_c), 0, 1);
      if (!NGUARD)
        __builtin_prefetch(B + (size_t)((it + 2) * BK + b_k) * ldb + tileN + b_c, 0, 1);
    }

    // ---- WMMA fragments from LDS ----
    // A 16x32 bf16 layout: lane m=lrow; elems 0..7 -> k = khalf*8 + e,
    //                      elems 8..15 -> k = 16 + khalf*8 + (e-8)
    v16bf afrag[2], bfrag[2];
#pragma unroll
    for (int i = 0; i < 2; ++i) {
      const int m = waveM + i * 16 + lrow;
      const v8bf lo = *(const v8bf*)&sA[cur][m][khalf * 8];
      const v8bf hi = *(const v8bf*)&sA[cur][m][khalf * 8 + 16];
#pragma unroll
      for (int e = 0; e < 8; ++e) { afrag[i][e] = lo[e]; afrag[i][e + 8] = hi[e]; }
    }
    // B 32x16 bf16 layout: lane n=lrow; elems 0..7 -> k = khalf*8 + e,
    //                      elems 8..15 -> k = 16 + khalf*8 + (e-8)
#pragma unroll
    for (int j = 0; j < 2; ++j) {
      const int n = waveN + j * 16 + lrow;
      const v8bf lo = *(const v8bf*)&sBt[cur][n][khalf * 8];
      const v8bf hi = *(const v8bf*)&sBt[cur][n][khalf * 8 + 16];
#pragma unroll
      for (int e = 0; e < 8; ++e) { bfrag[j][e] = lo[e]; bfrag[j][e + 8] = hi[e]; }
    }
#pragma unroll
    for (int i = 0; i < 2; ++i)
#pragma unroll
      for (int j = 0; j < 2; ++j)
        acc[i][j] = __builtin_amdgcn_wmma_f32_16x16x32_bf16(
            false, afrag[i], false, bfrag[j], (short)0, acc[i][j], false, false);

    if (hasNext) {
      commit_tile(cur ^ 1);   // writes go to the other buffer: no read/write hazard
      __syncthreads();        // one barrier per K-step
    }
  }

  // ---- epilogue + store; C/D layout: VGPR t -> row khalf*8+t, col lrow ----
  const size_t plane = (size_t)M * (size_t)ldc;   // conv_states plane stride (U*DIN)
#pragma unroll
  for (int i = 0; i < 2; ++i) {
#pragma unroll
    for (int j = 0; j < 2; ++j) {
      const int gn = tileN + waveN + j * 16 + lrow;
      if (NGUARD && gn >= N) continue;
      const int gm0 = tileM + waveM + i * 16 + khalf * 8;
#pragma unroll
      for (int t = 0; t < 8; ++t) {
        const int gm = gm0 + t;
        float v = acc[i][j][t];
        if (EPI == EPI_CONV_SILU) {
          const size_t idx = (size_t)gm * ldc + gn;
          const float c = ep1[0 * ldc + gn] * ep0[0 * plane + idx]
                        + ep1[1 * ldc + gn] * ep0[1 * plane + idx]
                        + ep1[2 * ldc + gn] * ep0[2 * plane + idx]
                        + ep1[3 * ldc + gn] * v
                        + ep2[gn];
          v = c * (1.0f / (1.0f + __expf(-c)));            // SiLU
        } else if (EPI == EPI_SOFTPLUS) {
          const float z = v + ep0[gn];
          v = fmaxf(z, 0.0f) + log1pf(__expf(-fabsf(z)));  // stable softplus
        }
        C[(size_t)gm * ldc + gn] = v;
      }
    }
  }
}

// Pointwise SSM state update: one thread per (u, d); streams ssm_state once.
// g[u,d] = (sum_n (ssm_state*exp(dt*A) + dt*h*B_n) * C_n + D_skip*h) * res
__global__ __launch_bounds__(256) void ssm_update_kernel(
    const float* __restrict__ dt, const float* __restrict__ h,
    const float* __restrict__ res, const float* __restrict__ dbc,
    const float* __restrict__ ssm_state, const float* __restrict__ A_log,
    const float* __restrict__ D_skip, float* __restrict__ g)
{
  __shared__ float sB[N_DIM];
  __shared__ float sC[N_DIM];
  const int u = blockIdx.y;
  const int d = blockIdx.x * blockDim.x + threadIdx.x;
  if (threadIdx.x < 2 * N_DIM) {
    const float val = dbc[(size_t)u * DBC_W + R_DIM + threadIdx.x];
    if (threadIdx.x < N_DIM) sB[threadIdx.x] = val;
    else                     sC[threadIdx.x - N_DIM] = val;
  }
  __syncthreads();

  const size_t ud = (size_t)u * DIN_DIM + d;
  const float dtv = dt[ud];
  const float hv  = h[ud];
  const float dth = dtv * hv;
  const float4* st = (const float4*)(ssm_state + ud * N_DIM);
  const float4* al = (const float4*)(A_log + (size_t)d * N_DIM);

  float acc = 0.f;
#pragma unroll
  for (int q = 0; q < 4; ++q) {
    const float4 s4 = st[q];
    const float4 a4 = al[q];
    const float sv[4] = {s4.x, s4.y, s4.z, s4.w};
    const float av[4] = {a4.x, a4.y, a4.z, a4.w};
#pragma unroll
    for (int r = 0; r < 4; ++r) {
      const int n = q * 4 + r;
      const float dA = __expf(-dtv * __expf(av[r]));   // exp(dt * (-exp(A_log)))
      const float ns = sv[r] * dA + dth * sB[n];
      acc += ns * sC[n];
    }
  }
  g[ud] = (acc + D_skip[d] * hv) * res[ud];
}

extern "C" void kernel_launch(void* const* d_in, const int* in_sizes, int n_in,
                              void* d_out, int out_size, void* d_ws, size_t ws_size,
                              hipStream_t stream) {
  (void)in_sizes; (void)n_in; (void)out_size; (void)ws_size;
  const float* x           = (const float*)d_in[0];   // (U, H)
  const float* conv_states = (const float*)d_in[1];   // (3, U, DIN)
  const float* ssm_state   = (const float*)d_in[2];   // (U, DIN, N)
  const float* ssm_proj    = (const float*)d_in[3];   // (H, DIN)
  const float* mlp_proj    = (const float*)d_in[4];   // (H, DIN)
  const float* down_proj   = (const float*)d_in[5];   // (DIN, H)
  const float* conv_w      = (const float*)d_in[6];   // (4, DIN)
  const float* conv_bias   = (const float*)d_in[7];   // (DIN)
  const float* x_proj_w    = (const float*)d_in[8];   // (DIN, 160)
  const float* dt_proj_w   = (const float*)d_in[9];   // (R, DIN)
  const float* dt_bias     = (const float*)d_in[10];  // (DIN)
  const float* A_log       = (const float*)d_in[11];  // (DIN, N)
  const float* D_skip      = (const float*)d_in[12];  // (DIN)
  float* out = (float*)d_out;                         // (U, H)

  float* ws      = (float*)d_ws;
  float* h_buf   = ws;                                        // U*DIN
  float* res_buf = h_buf  + (size_t)U_DIM * DIN_DIM;          // U*DIN
  float* dbc_buf = res_buf + (size_t)U_DIM * DIN_DIM;         // U*160
  float* dt_buf  = dbc_buf + (size_t)U_DIM * DBC_W;           // U*DIN
  float* g_buf   = dt_buf + (size_t)U_DIM * DIN_DIM;          // U*DIN

  const dim3 blk(256);

  // 1) h = silu(conv_w0..2 * conv_states + conv_w3 * (x @ ssm_proj) + bias)
  gemm_bf16_wmma<EPI_CONV_SILU, false><<<dim3(DIN_DIM / BN, U_DIM / BM), blk, 0, stream>>>(
      x, H_DIM, ssm_proj, DIN_DIM, h_buf, DIN_DIM,
      U_DIM, DIN_DIM, H_DIM, conv_states, conv_w, conv_bias);

  // 2) res = x @ mlp_proj
  gemm_bf16_wmma<EPI_NONE, false><<<dim3(DIN_DIM / BN, U_DIM / BM), blk, 0, stream>>>(
      x, H_DIM, mlp_proj, DIN_DIM, res_buf, DIN_DIM,
      U_DIM, DIN_DIM, H_DIM, nullptr, nullptr, nullptr);

  // 3) dbc = h @ x_proj_w   (N = 160: the only launch that pays bounds guards)
  gemm_bf16_wmma<EPI_NONE, true><<<dim3((DBC_W + BN - 1) / BN, U_DIM / BM), blk, 0, stream>>>(
      h_buf, DIN_DIM, x_proj_w, DBC_W, dbc_buf, DBC_W,
      U_DIM, DBC_W, DIN_DIM, nullptr, nullptr, nullptr);

  // 4) dt = softplus(dbc[:, :128] @ dt_proj_w + dt_bias)
  gemm_bf16_wmma<EPI_SOFTPLUS, false><<<dim3(DIN_DIM / BN, U_DIM / BM), blk, 0, stream>>>(
      dbc_buf, DBC_W, dt_proj_w, DIN_DIM, dt_buf, DIN_DIM,
      U_DIM, DIN_DIM, R_DIM, dt_bias, nullptr, nullptr);

  // 5) pointwise SSM state update -> g = y * res
  ssm_update_kernel<<<dim3(DIN_DIM / 256, U_DIM), blk, 0, stream>>>(
      dt_buf, h_buf, res_buf, dbc_buf, ssm_state, A_log, D_skip, g_buf);

  // 6) out = g @ down_proj
  gemm_bf16_wmma<EPI_NONE, false><<<dim3(H_DIM / BN, U_DIM / BM), blk, 0, stream>>>(
      g_buf, DIN_DIM, down_proj, H_DIM, out, H_DIM,
      U_DIM, H_DIM, DIN_DIM, nullptr, nullptr, nullptr);
}
